// FLARetNet_5351529251274
// MI455X (gfx1250) — matrix operand order repaired and checked by reading, not compiled, not run on hardware
//
#include <hip/hip_runtime.h>
#include <hip/hip_bf16.h>
#include <math.h>

typedef __attribute__((ext_vector_type(16))) __bf16 v16bf;
typedef __attribute__((ext_vector_type(4)))  __bf16 v4bf;
typedef __attribute__((ext_vector_type(8)))  float  v8f;
typedef __attribute__((ext_vector_type(4)))  float  v4f;

namespace {
constexpr int Bb = 4, Tt = 2048, Hh = 8;
constexpr int DK = 128, DV = 256, Cc = 64;
constexpr int NCHUNK = Tt / Cc;          // 32
constexpr int SLICES = 4;                // DV split: 4 x 64 columns
constexpr int SL = DV / SLICES;          // 64

// retention LDS layout (bytes); all offsets & row strides 16B-aligned for
// async b128 copies.
constexpr int QS = DK + 8;               // bf16 row stride (136 elem = 272 B)
constexpr int VS = SL + 8;               // 72 elem = 144 B
constexpr int AS = Cc + 8;               // 72
constexpr int S_OFF  = 0;                                    // float [128][64]
constexpr int Q_OFF  = S_OFF  + DK * SL * 4;                 // 32768
constexpr int K_OFF  = Q_OFF  + Cc * QS * 2;
constexpr int V_OFF  = K_OFF  + Cc * QS * 2;
constexpr int VD_OFF = V_OFF  + Cc * VS * 2;
constexpr int A_OFF  = VD_OFF + Cc * VS * 2;
constexpr int RET_SMEM = A_OFF + Cc * AS * 2;                // 95232 B
}

__device__ __forceinline__ int lane_id() { return (int)(threadIdx.x & 31u); }

__device__ __forceinline__ v8f wmma_bf16(v16bf a, v16bf b, v8f c) {
  return __builtin_amdgcn_wmma_f32_16x16x32_bf16(false, a, false, b, (short)0, c,
                                                 false, false);
}

// ---- CDNA5 async global->LDS (ASYNCcnt-tracked DMA path) ------------------
__device__ __forceinline__ void async_ld16(const __bf16* lds,
                                           const __bf16* gsrc) {
  unsigned l = (unsigned)(unsigned long long)lds;  // AS(3) offset = addr[31:0]
  asm volatile("global_load_async_to_lds_b128 %0, %1, off"
               :: "v"(l), "v"(gsrc) : "memory");
}
__device__ __forceinline__ void wait_async0() {
  asm volatile("s_wait_asynccnt 0x0" ::: "memory");
}

// ---- CDNA5 wave32 fragment loaders (contiguous k = 16h + s in A and B) -------------------
__device__ __forceinline__ v16bf load_a(const __bf16* p, int ld) {
  int l = lane_id(); int m = l & 15; int kh = (l >> 4) << 4;
  v16bf r;
#pragma unroll
  for (int v = 0; v < 8; ++v) {
    int k = 2 * v + kh;
    r[2 * v]     = p[m * ld + k];
    r[2 * v + 1] = p[m * ld + k + 1];
  }
  return r;
}
// memory holds A^T: element A[m][k] at p[k*ld + m]
__device__ __forceinline__ v16bf load_a_t(const __bf16* p, int ld) {
  int l = lane_id(); int m = l & 15; int kh = (l >> 4) << 4;
  v16bf r;
#pragma unroll
  for (int v = 0; v < 8; ++v) {
    int k = 2 * v + kh;
    r[2 * v]     = p[k * ld + m];
    r[2 * v + 1] = p[(k + 1) * ld + m];
  }
  return r;
}
__device__ __forceinline__ v16bf load_b(const __bf16* p, int ld) {
  int l = lane_id(); int n = l & 15; int kh = (l >> 4) << 4;
  v16bf r;
#pragma unroll
  for (int v = 0; v < 8; ++v) {
    int k = 2 * v + kh;
    r[2 * v]     = p[k * ld + n];
    r[2 * v + 1] = p[(k + 1) * ld + n];
  }
  return r;
}
// memory holds B^T (N x K): element B[k][n] at p[n*ld + k]
__device__ __forceinline__ v16bf load_b_t(const __bf16* p, int ld) {
  int l = lane_id(); int n = l & 15; int kh = (l >> 4) << 4;
  v16bf r;
#pragma unroll
  for (int v = 0; v < 8; ++v) {
    int k = 2 * v + kh;
    r[2 * v]     = p[n * ld + k];
    r[2 * v + 1] = p[n * ld + k + 1];
  }
  return r;
}
__device__ __forceinline__ v16bf load_b_f32(const float* p, int ld) {
  int l = lane_id(); int n = l & 15; int kh = (l >> 4) << 4;
  v16bf r;
#pragma unroll
  for (int v = 0; v < 8; ++v) {
    int k = 2 * v + kh;
    r[2 * v]     = (__bf16)p[k * ld + n];
    r[2 * v + 1] = (__bf16)p[(k + 1) * ld + n];
  }
  return r;
}
template <typename T>
__device__ __forceinline__ void store_c_t(T* p, int ld, v8f acc) {
  int l = lane_id(); int n = l & 15; int mb = (l >> 4) << 3;
#pragma unroll
  for (int r = 0; r < 8; ++r) p[(mb + r) * ld + n] = (T)acc[r];
}
__device__ __forceinline__ v8f load_c(const float* p, int ld) {
  int l = lane_id(); int n = l & 15; int mb = (l >> 4) << 3;
  v8f r;
#pragma unroll
  for (int rr = 0; rr < 8; ++rr) r[rr] = p[(mb + rr) * ld + n];
  return r;
}

// ---------------------------------------------------------------------------
// Vectorized f32 -> bf16 conversion (one-time; exact grid, no bounds check).
__global__ __launch_bounds__(256) void f32_to_bf16_kernel(
    const float* __restrict__ in, __bf16* __restrict__ out) {
  size_t i = (size_t)blockIdx.x * 256 + threadIdx.x;
  v4f f = ((const v4f*)in)[i];
  v4bf o;
  o[0] = (__bf16)f[0]; o[1] = (__bf16)f[1];
  o[2] = (__bf16)f[2]; o[3] = (__bf16)f[3];
  ((v4bf*)out)[i] = o;
}

// ---------------------------------------------------------------------------
// GEMM: C(MxN) = A(bf16, MxK) @ B(bf16, KxN). Block tile 128x256, k-step 32,
// 256 threads = 8 waves (2x4), wave tile 64x64 (16 WMMAs / 8 frag loads).
// LDS double-buffered; staging uses async global->LDS b128 (no VALU).
template <typename OutT>
__global__ __launch_bounds__(256) void gemm_async_kernel(
    const __bf16* __restrict__ Ag, const __bf16* __restrict__ Bg,
    OutT* __restrict__ Cg, int N, int K) {
  __shared__ __bf16 As[2][128 * 40];   // 128x32, row stride 40 (80 B, 16B-mult)
  __shared__ __bf16 Bs[2][32 * 264];   // 32x256, row stride 264 (528 B)
  const int bm = blockIdx.x * 128;
  const int bn = blockIdx.y * 256;
  const int tid = (int)threadIdx.x;
  const int w = tid >> 5;
  const int wm = (w >> 2) * 64;
  const int wn = (w & 3) * 64;

  v8f acc[4][4];
#pragma unroll
  for (int i = 0; i < 4; ++i)
#pragma unroll
    for (int j = 0; j < 4; ++j)
#pragma unroll
      for (int z = 0; z < 8; ++z) acc[i][j][z] = 0.f;

  auto stage = [&](int buf, int k0) {
    // A: 512 x 16B chunks
#pragma unroll
    for (int id = tid; id < 512; id += 256) {
      int r = id >> 2, c = id & 3;
      async_ld16(&As[buf][r * 40 + c * 8],
                 Ag + (size_t)(bm + r) * K + k0 + c * 8);
    }
    // B: 1024 x 16B chunks
#pragma unroll
    for (int id = tid; id < 1024; id += 256) {
      int r = id >> 5, c = id & 31;
      async_ld16(&Bs[buf][r * 264 + c * 8],
                 Bg + (size_t)(k0 + r) * N + bn + c * 8);
    }
  };

  stage(0, 0);
  wait_async0();
  __syncthreads();

  const int nk = K / 32;
  for (int kk = 0; kk < nk; ++kk) {
    const int cur = kk & 1;
    if (kk + 1 < nk) stage(cur ^ 1, (kk + 1) * 32);  // prefetch next tile
    v16bf af[4];
#pragma unroll
    for (int i = 0; i < 4; ++i)
      af[i] = load_a(&As[cur][(wm + 16 * i) * 40], 40);
#pragma unroll
    for (int j = 0; j < 4; ++j) {
      v16bf bfr = load_b(&Bs[cur][wn + 16 * j], 264);
#pragma unroll
      for (int i = 0; i < 4; ++i) acc[i][j] = wmma_bf16(af[i], bfr, acc[i][j]);
    }
    wait_async0();
    __syncthreads();
  }
#pragma unroll
  for (int i = 0; i < 4; ++i)
#pragma unroll
    for (int j = 0; j < 4; ++j)
      store_c_t(Cg + (size_t)(bm + wm + 16 * i) * N + (bn + wn + 16 * j), N,
                acc[i][j]);
}

// ---------------------------------------------------------------------------
// In-place RoPE on bf16 (B,T,H,DK); folds q-scaling into `scale`.
__global__ __launch_bounds__(256) void rope_kernel(__bf16* __restrict__ x,
                                                   float scale) {
  size_t idx = (size_t)blockIdx.x * 256 + threadIdx.x;  // one per pair
  int p = (int)(idx & 63);
  size_t r = idx >> 6;                                  // (b*T + t)*H + h
  int t = (int)((r / Hh) % Tt);
  float inv = __powf(10000.f, -(float)(2 * p) / (float)DK);
  float f = (float)t * inv;
  float cs = __cosf(f), sn = __sinf(f);
  __bf16* base = x + r * DK;
  float x1 = (float)base[p], x2 = (float)base[p + 64];
  base[p]      = (__bf16)((x1 * cs - x2 * sn) * scale);
  base[p + 64] = (__bf16)((x2 * cs + x1 * sn) * scale);
}

// ---------------------------------------------------------------------------
// Chunked retention scan. Grid (B*H, SLICES); 256 threads = 8 waves.
// State S (DK x SL, f32) lives in LDS across the 32-chunk scan; q/k/v chunks
// arrive via async global->LDS copies (already bf16, RoPE pre-applied).
// All decay factors are hoisted out of the scan loop (loop-invariant per lane).
__global__ __launch_bounds__(256) void retention_kernel(
    const __bf16* __restrict__ q, const __bf16* __restrict__ k,
    const __bf16* __restrict__ v, float* __restrict__ o) {
  const int bh = blockIdx.x;
  const int sl = blockIdx.y;
  const int b = bh / Hh, h = bh % Hh;
  const float gamma = 1.f - exp2f(-5.f - (float)h);
  const float lg = __logf(gamma);
  const float gC = __expf((float)Cc * lg);

  extern __shared__ char smem[];
  float*  Ssm  = (float*)(smem + S_OFF);
  __bf16* qsm  = (__bf16*)(smem + Q_OFF);
  __bf16* ksm  = (__bf16*)(smem + K_OFF);
  __bf16* vsm  = (__bf16*)(smem + V_OFF);
  __bf16* vdsm = (__bf16*)(smem + VD_OFF);
  __bf16* atn  = (__bf16*)(smem + A_OFF);

  const int tid = (int)threadIdx.x;
  const int w = tid >> 5;
  const int l = lane_id();
  const int cn = l & 15;
  const int cmb = (l >> 4) << 3;

  // ---- hoisted decay constants (invariant across all 32 chunks) ----
  float dkv[16];                        // dec_k for the vdsm rebuild slots
#pragma unroll
  for (int e = 0; e < 16; ++e) {
    int i = (tid + e * 256) >> 6;       // chunk-row this slot always touches
    dkv[e] = __expf((float)(Cc - 1 - i) * lg);
  }
  float mval[2][8], dqv[2][8];          // causal-decay mask & dec_q per tile
#pragma unroll
  for (int tt = 0; tt < 2; ++tt) {
    int tile = w * 2 + tt;
    int ti = tile >> 2, tj = tile & 3;
#pragma unroll
    for (int r = 0; r < 8; ++r) {
      int i = ti * 16 + cmb + r, j = tj * 16 + cn;
      mval[tt][r] = (i >= j) ? __expf((float)(i - j) * lg) : 0.f;
      dqv[tt][r]  = __expf((float)(i + 1) * lg);
    }
  }

  for (int i = tid; i < DK * SL; i += 256) Ssm[i] = 0.f;
  __syncthreads();

  for (int nc = 0; nc < NCHUNK; ++nc) {
    const int t0 = nc * Cc;
    // ---- async-stage q/k (64x128) and v slice (64x64), all bf16 ----
#pragma unroll
    for (int id = tid; id < Cc * (DK / 8); id += 256) {   // 1024 chunks
      int i = id >> 4, c = id & 15;
      size_t gb = ((size_t)(b * Tt + t0 + i) * Hh + h);
      async_ld16(qsm + i * QS + c * 8, q + gb * DK + c * 8);
      async_ld16(ksm + i * QS + c * 8, k + gb * DK + c * 8);
    }
#pragma unroll
    for (int id = tid; id < Cc * (SL / 8); id += 256) {   // 512 chunks
      int i = id >> 3, c = id & 7;
      size_t gb = ((size_t)(b * Tt + t0 + i) * Hh + h);
      async_ld16(vsm + i * VS + c * 8, v + gb * DV + sl * SL + c * 8);
    }
    wait_async0();
    __syncthreads();

    // ---- attn = (q @ k^T) * decay mask -> bf16 LDS; also build dec_k*v ----
#pragma unroll
    for (int e = 0; e < 16; ++e) {
      int idx = tid + e * 256;
      int i = idx >> 6, c = idx & 63;
      float vv = (float)vsm[i * VS + c];
      vdsm[i * VS + c] = (__bf16)(vv * dkv[e]);
    }
    for (int tt = 0; tt < 2; ++tt) {
      int tile = w * 2 + tt;
      int ti = tile >> 2, tj = tile & 3;
      v8f acc;
#pragma unroll
      for (int z = 0; z < 8; ++z) acc[z] = 0.f;
#pragma unroll
      for (int ks = 0; ks < DK / 32; ++ks) {
        v16bf a  = load_a(qsm + (ti * 16) * QS + ks * 32, QS);
        v16bf bt = load_b_t(ksm + (tj * 16) * QS + ks * 32, QS);
        acc = wmma_bf16(a, bt, acc);
      }
#pragma unroll
      for (int r = 0; r < 8; ++r) {
        int i = ti * 16 + cmb + r, j = tj * 16 + cn;
        atn[i * AS + j] = (__bf16)(acc[r] * mval[tt][r]);
      }
    }
    __syncthreads();

    // ---- o = dec_q * (q @ S) + attn @ v  (64 x SL) ----
    for (int tt = 0; tt < 2; ++tt) {
      int tile = w * 2 + tt;
      int ti = tile >> 2, tj = tile & 3;
      v8f acc;
#pragma unroll
      for (int z = 0; z < 8; ++z) acc[z] = 0.f;
#pragma unroll
      for (int ks = 0; ks < DK / 32; ++ks) {
        v16bf a  = load_a(qsm + (ti * 16) * QS + ks * 32, QS);
        v16bf bs = load_b_f32(Ssm + (ks * 32) * SL + tj * 16, SL);
        acc = wmma_bf16(a, bs, acc);
      }
#pragma unroll
      for (int r = 0; r < 8; ++r) acc[r] *= dqv[tt][r];
#pragma unroll
      for (int ks = 0; ks < Cc / 32; ++ks) {
        v16bf a  = load_a(atn + (ti * 16) * AS + ks * 32, AS);
        v16bf bv = load_b(vsm + (ks * 32) * VS + tj * 16, VS);
        acc = wmma_bf16(a, bv, acc);
      }
#pragma unroll
      for (int r = 0; r < 8; ++r) {
        size_t gb = ((size_t)(b * Tt + t0 + ti * 16 + cmb + r) * Hh + h);
        o[gb * DV + sl * SL + tj * 16 + cn] = acc[r];
      }
    }
    __syncthreads();   // all S reads done before the update below

    // ---- S = gamma^C * S + k^T @ (dec_k * v)  (DK x SL) ----
    for (int tt = 0; tt < 4; ++tt) {
      int tile = w * 4 + tt;
      int ti = tile >> 2, tj = tile & 3;
      float* sp = Ssm + (ti * 16) * SL + tj * 16;
      v8f acc = load_c(sp, SL);
#pragma unroll
      for (int z = 0; z < 8; ++z) acc[z] *= gC;
#pragma unroll
      for (int ks = 0; ks < Cc / 32; ++ks) {
        v16bf a  = load_a_t(ksm + (ks * 32) * QS + ti * 16, QS);
        v16bf bv = load_b(vdsm + (ks * 32) * VS + tj * 16, VS);
        acc = wmma_bf16(a, bv, acc);
      }
      store_c_t(sp, SL, acc);
    }
    __syncthreads();
  }
}

// ---------------------------------------------------------------------------
// RMS-norm over DV + SiLU gate; emits bf16 for the output GEMM.
__global__ __launch_bounds__(256) void gate_kernel(const float* __restrict__ oi,
                                                   const float* __restrict__ g,
                                                   const float* __restrict__ nw,
                                                   __bf16* __restrict__ oo) {
  const size_t row = blockIdx.x;
  const int c = (int)threadIdx.x;
  float x = oi[row * DV + c];
  __shared__ float red[256];
  red[c] = x * x;
  __syncthreads();
  for (int s = 128; s > 0; s >>= 1) {
    if (c < s) red[c] += red[c + s];
    __syncthreads();
  }
  float inv_rms = rsqrtf(red[0] / (float)DV + 1e-5f);
  float gv = g[row * DV + c];
  float sig = 1.f / (1.f + __expf(-gv));
  oo[row * DV + c] = (__bf16)(x * inv_rms * nw[c] * gv * sig);
}

// ---------------------------------------------------------------------------
extern "C" void kernel_launch(void* const* d_in, const int* in_sizes, int n_in,
                              void* d_out, int out_size, void* d_ws,
                              size_t ws_size, hipStream_t stream) {
  (void)in_sizes; (void)n_in; (void)out_size; (void)ws_size;
  const float* x  = (const float*)d_in[0];
  const float* Wq = (const float*)d_in[1];
  const float* Wk = (const float*)d_in[2];
  const float* Wv = (const float*)d_in[3];
  const float* Wg = (const float*)d_in[4];
  const float* Wo = (const float*)d_in[5];
  const float* nw = (const float*)d_in[6];
  float* out = (float*)d_out;

  const size_t MT = (size_t)Bb * Tt;  // 8192 rows
  char* wsb = (char*)d_ws;            // total use: 224 MiB
  __bf16* xb  = (__bf16*)(wsb);                    // 16 MiB (8192x1024)
  __bf16* wqb = (__bf16*)(wsb + (16u << 20));      // 2 MiB
  __bf16* wkb = (__bf16*)(wsb + (18u << 20));      // 2 MiB
  __bf16* wvb = (__bf16*)(wsb + (20u << 20));      // 4 MiB
  __bf16* wgb = (__bf16*)(wsb + (24u << 20));      // 4 MiB
  __bf16* wob = (__bf16*)(wsb + (28u << 20));      // 4 MiB
  __bf16* qb  = (__bf16*)(wsb + (32u << 20));      // 16 MiB (8192x1024)
  __bf16* kb  = (__bf16*)(wsb + (48u << 20));      // 16 MiB
  __bf16* vb  = (__bf16*)(wsb + (64u << 20));      // 32 MiB (8192x2048)
  float*  gf  = (float*) (wsb + (96u << 20));      // 64 MiB (8192x2048 f32)
  float*  of  = (float*) (wsb + (160u << 20));     // 64 MiB
  __bf16* ogb = qb;  // q/k dead after retention; 32 MiB overlay for gated o

  dim3 blk(256);
  // one-time f32 -> bf16 conversions (vectorized x4)
  f32_to_bf16_kernel<<<dim3(8192), blk, 0, stream>>>(x, xb);
  f32_to_bf16_kernel<<<dim3(1024), blk, 0, stream>>>(Wq, wqb);
  f32_to_bf16_kernel<<<dim3(1024), blk, 0, stream>>>(Wk, wkb);
  f32_to_bf16_kernel<<<dim3(2048), blk, 0, stream>>>(Wv, wvb);
  f32_to_bf16_kernel<<<dim3(2048), blk, 0, stream>>>(Wg, wgb);
  f32_to_bf16_kernel<<<dim3(2048), blk, 0, stream>>>(Wo, wob);

  // projections (async-staged bf16 WMMA GEMMs, 128x256 tiles)
  gemm_async_kernel<__bf16><<<dim3(64, 4), blk, 0, stream>>>(xb, wqb, qb,
                                                             1024, 1024);
  gemm_async_kernel<__bf16><<<dim3(64, 4), blk, 0, stream>>>(xb, wkb, kb,
                                                             1024, 1024);
  gemm_async_kernel<__bf16><<<dim3(64, 8), blk, 0, stream>>>(xb, wvb, vb,
                                                             2048, 1024);
  gemm_async_kernel<float><<<dim3(64, 8), blk, 0, stream>>>(xb, wgb, gf,
                                                            2048, 1024);

  // RoPE in place on bf16 (q gets DK^-0.5 folded in)
  const unsigned rope_blocks = (unsigned)((MT * Hh * (DK / 2)) / 256);
  rope_kernel<<<dim3(rope_blocks), blk, 0, stream>>>(qb,
                                                     1.0f / sqrtf((float)DK));
  rope_kernel<<<dim3(rope_blocks), blk, 0, stream>>>(kb, 1.0f);

  // chunked retention scan (S in LDS, 95 KB dynamic)
  (void)hipFuncSetAttribute((const void*)retention_kernel,
                            hipFuncAttributeMaxDynamicSharedMemorySize,
                            RET_SMEM);
  retention_kernel<<<dim3(Bb * Hh, SLICES), blk, RET_SMEM, stream>>>(qb, kb, vb,
                                                                     of);

  // RMS norm + SiLU gate -> bf16
  gate_kernel<<<dim3((unsigned)(MT * Hh)), blk, 0, stream>>>(of, gf, nw, ogb);

  // output projection
  gemm_async_kernel<float><<<dim3(64, 4), blk, 0, stream>>>(ogb, wob, out,
                                                            1024, 2048);
}
